// KVCacheAttention_63230508531958
// MI455X (gfx1250) — matrix operand order repaired and checked by reading, not compiled, not run on hardware
//
#include <hip/hip_runtime.h>

typedef __attribute__((ext_vector_type(16))) __bf16 v16bf;
typedef __attribute__((ext_vector_type(8)))  float  v8f;

constexpr int T  = 4096;   // sequence length
constexpr int DM = 1024;   // d_model
constexpr int NH = 16;     // heads
constexpr int DH = 64;     // head dim

// ---------------- helpers ----------------

__device__ inline unsigned short f32_to_bf16(float f) {
    unsigned int u = __builtin_bit_cast(unsigned int, f);
    u += 0x7FFFu + ((u >> 16) & 1u);           // round-to-nearest-even
    return (unsigned short)(u >> 16);
}

// gfx1250 async global->LDS copy, 16B per lane, tracked by ASYNCcnt.
__device__ inline void async_copy_b128(unsigned lds_addr, const void* gaddr) {
    asm volatile("global_load_async_to_lds_b128 %0, %1, off"
                 :: "v"(lds_addr),
                    "v"((unsigned long long)(uintptr_t)gaddr)
                 : "memory");
}
__device__ inline void wait_asynccnt_8() { asm volatile("s_wait_asynccnt 0x8" ::: "memory"); }
__device__ inline void wait_asynccnt_4() { asm volatile("s_wait_asynccnt 0x4" ::: "memory"); }
__device__ inline void wait_asynccnt_0() { asm volatile("s_wait_asynccnt 0x0" ::: "memory"); }

__device__ inline unsigned lds_off(const void* p) {
    return (unsigned)(uintptr_t)p;             // generic LDS addr: low 32 bits = LDS offset
}

// A-fragment, 16x32 (MxK) bf16, ISA 7.12.2 layout:
// lane L: row = mbase + L%16 ; kh = L/16
// VGPR j (j<4): K = kbase + kh*8 + 2j,2j+1 ; VGPR j+4: K = kbase + 16 + kh*8 + 2j,2j+1
__device__ inline v16bf load_afrag(const unsigned short* src, int ld,
                                   int mbase, int kbase, int lane) {
    const int m  = mbase + (lane & 15);
    const int kh = (lane >> 4) & 1;
    const unsigned short* row = src + (size_t)m * ld + kbase;
    union { unsigned int u[8]; v16bf v; } f;
#pragma unroll
    for (int j = 0; j < 4; ++j) {
        f.u[j]     = *(const unsigned int*)(row + kh * 8 + 2 * j);
        f.u[j + 4] = *(const unsigned int*)(row + 16 + kh * 8 + 2 * j);
    }
    return f.v;
}

// B-fragment, 32x16 (KxN) bf16. Source row-major indexed [n][k].
// lane L: col n = nbase + L%16 ; kh = L/16 ; VGPR j (j<4): K = kbase + kh*8 + 2j,2j+1 ; VGPR j+4: K = kbase + 16 + kh*8 + 2j,2j+1
__device__ inline v16bf load_bfrag(const unsigned short* src, int ld,
                                   int nbase, int kbase, int lane) {
    const int n  = nbase + (lane & 15);
    const int kh = (lane >> 4) & 1;
    const unsigned short* p = src + (size_t)n * ld + kbase;
    union { unsigned int u[8]; v16bf v; } f;
#pragma unroll
    for (int j = 0; j < 4; ++j) {
        f.u[j]     = *(const unsigned int*)(p + kh * 8 + 2 * j);
        f.u[j + 4] = *(const unsigned int*)(p + 16 + kh * 8 + 2 * j);
    }
    return f.v;
}

__device__ inline v8f wmma_bf16(v16bf a, v16bf b, v8f c) {
    return __builtin_amdgcn_wmma_f32_16x16x32_bf16(
        false, a, false, b, (short)0, c, false, false);
}

// ---------------- fp32 -> bf16 convert ----------------

__global__ __launch_bounds__(256) void cvt_f32_bf16(const float* __restrict__ in,
                                                    unsigned short* __restrict__ out,
                                                    int n) {
    int i = blockIdx.x * 256 + threadIdx.x;
    if (i < n) out[i] = f32_to_bf16(in[i]);
}

// ---------------- GEMM: Y[m,n] = sum_k A[m,k]*B[n,k] + bias[n] ----------------
// A: bf16 [4096][1024] row-major, B: bf16 [1024][1024] row-major (nn.Linear W).
// MODE 0: out bf16 [H][T][DH]   (Q, K)
// MODE 1: out bf16 [H][DH][T]   (V transposed per head, for PV B-operand)
// MODE 2: out fp32 [T][DM]      (final projection)
template <int MODE>
__global__ __launch_bounds__(128) void gemm_bf16(const unsigned short* __restrict__ A,
                                                 const unsigned short* __restrict__ B,
                                                 const float* __restrict__ bias,
                                                 void* __restrict__ outp) {
    __shared__ __align__(16) unsigned short As[2][64][32];
    __shared__ __align__(16) unsigned short Bs[2][64][32];

    const int tid  = threadIdx.x;
    const int w    = tid >> 5;
    const int lane = tid & 31;
    const int m0   = blockIdx.y * 64;
    const int n0   = blockIdx.x * 64;

    // async double-buffered tile fetch: per thread 2 b128 for A + 2 for B
    auto issue = [&](int buf, int k0) {
#pragma unroll
        for (int i = 0; i < 2; ++i) {
            int u = tid + i * 128;               // 0..255 -> 64 rows x 4 chunks
            int row = u >> 2, c8 = (u & 3) * 8;
            async_copy_b128(lds_off(&As[buf][row][c8]),
                            A + (size_t)(m0 + row) * DM + k0 + c8);
            async_copy_b128(lds_off(&Bs[buf][row][c8]),
                            B + (size_t)(n0 + row) * DM + k0 + c8);
        }
    };

    const v8f zero = {0.f, 0.f, 0.f, 0.f, 0.f, 0.f, 0.f, 0.f};
    v8f acc[4] = {zero, zero, zero, zero};

    issue(0, 0);
    for (int j = 0; j < DM / 32; ++j) {
        const int buf = j & 1;
        if (j + 1 < DM / 32) { issue(buf ^ 1, (j + 1) * 32); wait_asynccnt_4(); }
        else                 { wait_asynccnt_0(); }
        __syncthreads();
        v16bf a = load_afrag(&As[buf][0][0], 32, w * 16, 0, lane);
#pragma unroll
        for (int nt = 0; nt < 4; ++nt) {
            v16bf b = load_bfrag(&Bs[buf][0][0], 32, nt * 16, 0, lane);
            acc[nt] = wmma_bf16(a, b, acc[nt]);
        }
        __syncthreads();                         // protect buf before its refill
    }

#pragma unroll
    for (int nt = 0; nt < 4; ++nt) {
        const int n  = n0 + nt * 16 + (lane & 15);
        const float bv = bias[n];
#pragma unroll
        for (int r = 0; r < 8; ++r) {
            const int m = m0 + w * 16 + r + 8 * (lane >> 4);
            const float val = acc[nt][r] + bv;
            if (MODE == 0) {
                const int h = n >> 6, d = n & 63;
                ((unsigned short*)outp)[(size_t)h * T * DH + (size_t)m * DH + d] = f32_to_bf16(val);
            } else if (MODE == 1) {
                const int h = n >> 6, d = n & 63;
                ((unsigned short*)outp)[(size_t)h * DH * T + (size_t)d * T + m] = f32_to_bf16(val);
            } else {
                ((float*)outp)[(size_t)m * DM + n] = val;
            }
        }
    }
}

// ---------------- flash attention (one head, 64 queries per block) ----------------
// Q,K: bf16 [H][T][DH]; Vt: bf16 [H][DH][T]; O: bf16 [T][DM]
__global__ __launch_bounds__(128) void flash_attn(const unsigned short* __restrict__ Q,
                                                  const unsigned short* __restrict__ K,
                                                  const unsigned short* __restrict__ Vt,
                                                  unsigned short* __restrict__ O) {
    __shared__ __align__(16) unsigned short Ks[2][64][64];   // K tile  [key][dim]
    __shared__ __align__(16) unsigned short Vs[2][64][64];   // V^T tile [dim][key]
    __shared__ __align__(16) unsigned short Pst[4][16][64];  // wave-private P staging

    const int tid  = threadIdx.x;
    const int w    = tid >> 5;
    const int lane = tid & 31;
    const int h    = blockIdx.y;
    const int q0   = blockIdx.x * 64 + w * 16;

    const unsigned short* Qh = Q  + (size_t)h * T * DH;
    const unsigned short* Kh = K  + (size_t)h * T * DH;
    const unsigned short* Vh = Vt + (size_t)h * DH * T;
    unsigned short* Pw = &Pst[w][0][0];

    // async double-buffered K/V tiles: 512 b128 per tile pair half => 4+4 per thread
    auto issue_tile = [&](int buf, int kt0) {
#pragma unroll
        for (int i = 0; i < 4; ++i) {
            int u = tid + i * 128;               // 0..511 -> 64 rows x 8 chunks
            int row = u >> 3, c = (u & 7) * 8;
            async_copy_b128(lds_off(&Ks[buf][row][c]),
                            Kh + (size_t)(kt0 + row) * DH + c);
        }
#pragma unroll
        for (int i = 0; i < 4; ++i) {
            int u = tid + i * 128;
            int row = u >> 3, c = (u & 7) * 8;
            async_copy_b128(lds_off(&Vs[buf][row][c]),
                            Vh + (size_t)row * T + kt0 + c);
        }
    };

    // Q strip 16x64 held in registers for the whole key sweep
    const v16bf qa0 = load_afrag(Qh, DH, q0, 0, lane);
    const v16bf qa1 = load_afrag(Qh, DH, q0, 32, lane);

    const v8f zero = {0.f, 0.f, 0.f, 0.f, 0.f, 0.f, 0.f, 0.f};
    v8f oacc[4] = {zero, zero, zero, zero};
    float mrow[8], lrow[8];
#pragma unroll
    for (int r = 0; r < 8; ++r) { mrow[r] = -3.0e38f; lrow[r] = 0.f; }

    constexpr int NTILES = T / 64;
    issue_tile(0, 0);
    for (int j = 0; j < NTILES; ++j) {
        const int buf = j & 1;
        if (j + 1 < NTILES) { issue_tile(buf ^ 1, (j + 1) * 64); wait_asynccnt_8(); }
        else                { wait_asynccnt_0(); }
        __syncthreads();

        // S = Q K^T  (16 x 64 per wave), K fragments from LDS (shared by 4 waves)
        v8f s[4] = {zero, zero, zero, zero};
#pragma unroll
        for (int nt = 0; nt < 4; ++nt) {
            v16bf b0 = load_bfrag(&Ks[buf][0][0], 64, nt * 16, 0, lane);
            s[nt] = wmma_bf16(qa0, b0, s[nt]);
            v16bf b1 = load_bfrag(&Ks[buf][0][0], 64, nt * 16, 32, lane);
            s[nt] = wmma_bf16(qa1, b1, s[nt]);
        }
#pragma unroll
        for (int nt = 0; nt < 4; ++nt) s[nt] *= 0.125f;   // 1/sqrt(DH)

        // online softmax (xor 1..8 keeps the two 16-lane halves independent)
        float mnew[8];
#pragma unroll
        for (int r = 0; r < 8; ++r) {
            float t = fmaxf(fmaxf(s[0][r], s[1][r]), fmaxf(s[2][r], s[3][r]));
            t = fmaxf(t, __shfl_xor(t, 1, 32));
            t = fmaxf(t, __shfl_xor(t, 2, 32));
            t = fmaxf(t, __shfl_xor(t, 4, 32));
            t = fmaxf(t, __shfl_xor(t, 8, 32));
            mnew[r] = fmaxf(mrow[r], t);
        }
#pragma unroll
        for (int r = 0; r < 8; ++r) {
            const float corr = __expf(mrow[r] - mnew[r]);
            float rs = 0.f;
#pragma unroll
            for (int nt = 0; nt < 4; ++nt) {
                const float p = __expf(s[nt][r] - mnew[r]);
                s[nt][r] = p;
                rs += p;
            }
            rs += __shfl_xor(rs, 1, 32);
            rs += __shfl_xor(rs, 2, 32);
            rs += __shfl_xor(rs, 4, 32);
            rs += __shfl_xor(rs, 8, 32);
            lrow[r] = lrow[r] * corr + rs;
            mrow[r] = mnew[r];
#pragma unroll
            for (int nt = 0; nt < 4; ++nt) oacc[nt][r] *= corr;
        }

        // stage P (C/D layout -> row-major bf16) in wave-private LDS
#pragma unroll
        for (int nt = 0; nt < 4; ++nt) {
            const int col = nt * 16 + (lane & 15);
#pragma unroll
            for (int r = 0; r < 8; ++r) {
                const int row = r + 8 * (lane >> 4);
                Pw[row * 64 + col] = f32_to_bf16(s[nt][r]);
            }
        }

        // O += P x V   (V^T fragments from LDS: contiguous key pairs)
#pragma unroll
        for (int k0 = 0; k0 < 64; k0 += 32) {
            v16bf pa = load_afrag(Pw, 64, 0, k0, lane);
#pragma unroll
            for (int dt = 0; dt < 4; ++dt) {
                v16bf vb = load_bfrag(&Vs[buf][0][0], 64, dt * 16, k0, lane);
                oacc[dt] = wmma_bf16(pa, vb, oacc[dt]);
            }
        }
        __syncthreads();                         // protect buf before its refill
    }

    float invl[8];
#pragma unroll
    for (int r = 0; r < 8; ++r) invl[r] = 1.0f / lrow[r];
#pragma unroll
    for (int dt = 0; dt < 4; ++dt) {
        const int d = dt * 16 + (lane & 15);
#pragma unroll
        for (int r = 0; r < 8; ++r) {
            const int m = q0 + r + 8 * (lane >> 4);
            O[(size_t)m * DM + h * DH + d] = f32_to_bf16(oacc[dt][r] * invl[r]);
        }
    }
}

// ---------------- launch ----------------

extern "C" void kernel_launch(void* const* d_in, const int* in_sizes, int n_in,
                              void* d_out, int out_size, void* d_ws, size_t ws_size,
                              hipStream_t stream) {
    const float* x  = (const float*)d_in[0];
    const float* Wq = (const float*)d_in[1];
    const float* bq = (const float*)d_in[2];
    const float* Wk = (const float*)d_in[3];
    const float* bk = (const float*)d_in[4];
    const float* Wv = (const float*)d_in[5];
    const float* bv = (const float*)d_in[6];
    const float* Wo = (const float*)d_in[7];
    const float* bo = (const float*)d_in[8];
    float* out = (float*)d_out;

    unsigned short* p   = (unsigned short*)d_ws;
    unsigned short* xb  = p; p += (size_t)T * DM;    // x  bf16 (8 MB)
    unsigned short* wqb = p; p += (size_t)DM * DM;   // weights bf16 (2 MB each)
    unsigned short* wkb = p; p += (size_t)DM * DM;
    unsigned short* wvb = p; p += (size_t)DM * DM;
    unsigned short* wob = p; p += (size_t)DM * DM;
    unsigned short* Qb  = p; p += (size_t)T * DM;    // [H][T][DH]
    unsigned short* Kb  = p; p += (size_t)T * DM;    // [H][T][DH]
    unsigned short* Vtb = p; p += (size_t)T * DM;    // [H][DH][T]
    unsigned short* Ob  = p; p += (size_t)T * DM;    // [T][DM]

    cvt_f32_bf16<<<(T * DM) / 256, 256, 0, stream>>>(x, xb, T * DM);
    cvt_f32_bf16<<<(DM * DM) / 256, 256, 0, stream>>>(Wq, wqb, DM * DM);
    cvt_f32_bf16<<<(DM * DM) / 256, 256, 0, stream>>>(Wk, wkb, DM * DM);
    cvt_f32_bf16<<<(DM * DM) / 256, 256, 0, stream>>>(Wv, wvb, DM * DM);
    cvt_f32_bf16<<<(DM * DM) / 256, 256, 0, stream>>>(Wo, wob, DM * DM);

    dim3 gg(DM / 64, T / 64);                        // (16, 64) blocks of 128 threads
    gemm_bf16<0><<<gg, 128, 0, stream>>>(xb, wqb, bq, Qb);
    gemm_bf16<0><<<gg, 128, 0, stream>>>(xb, wkb, bk, Kb);
    gemm_bf16<1><<<gg, 128, 0, stream>>>(xb, wvb, bv, Vtb);

    flash_attn<<<dim3(T / 64, NH), 128, 0, stream>>>(Qb, Kb, Vtb, Ob);

    gemm_bf16<2><<<gg, 128, 0, stream>>>(Ob, wob, bo, out);
}
